// VectorQuantizer_67379446939659
// MI455X (gfx1250) — matrix operand order repaired and checked by reading, not compiled, not run on hardware
//
#include <hip/hip_runtime.h>

typedef __attribute__((ext_vector_type(16))) _Float16 v16h;
typedef __attribute__((ext_vector_type(8)))  _Float16 v8h;
typedef __attribute__((ext_vector_type(4)))  _Float16 v4h;
typedef __attribute__((ext_vector_type(8)))  float    v8f;
typedef __attribute__((ext_vector_type(4)))  int      v4i;
typedef __attribute__((address_space(3)))    v4i      lds_v4i;

#define DIMS      256
#define NROWS     32768
#define KENT      8192
#define CB_CHUNK  64
#define NCHUNK    (KENT / CB_CHUNK)   // 128
#define NUMEL     8388608             // 8*256*64*64

#if __has_builtin(__builtin_amdgcn_global_load_async_to_lds_b128) && \
    __has_builtin(__builtin_amdgcn_s_wait_asynccnt)
#define VQ_ASYNC 1
#else
#define VQ_ASYNC 0
#endif

// Issue one 32KB codebook chunk (64 entries x 256 f16) into an LDS buffer.
// Each of the 256 threads moves 8 x 16B.  Async path: 8 ASYNCcnt ops per wave.
__device__ __forceinline__ void cb_issue_chunk(const _Float16* __restrict__ g,
                                               _Float16* l, int tid)
{
#if VQ_ASYNC
    #pragma unroll
    for (int i = 0; i < 8; ++i) {
        v4i*     gp = (v4i*)(g + (size_t)(tid + i * 256) * 8);
        lds_v4i* lp = (lds_v4i*)(l + (size_t)(tid + i * 256) * 8);
        __builtin_amdgcn_global_load_async_to_lds_b128(gp, lp, 0, 0);
    }
#else
    const float4* src = (const float4*)g;
    float4* dst = (float4*)l;
    #pragma unroll
    for (int i = 0; i < 8; ++i) dst[tid + i * 256] = src[tid + i * 256];
#endif
}

__device__ __forceinline__ void cb_wait_keep8()
{
#if VQ_ASYNC
    __builtin_amdgcn_s_wait_asynccnt(8);   // oldest chunk landed, next may fly
#endif
}

__device__ __forceinline__ void cb_wait_all()
{
#if VQ_ASYNC
    __builtin_amdgcn_s_wait_asynccnt(0);
#endif
}

// ---------------------------------------------------------------------------
// Kernel 1: codebook f32 -> f16, plus ||e||^2 per entry. grid=KENT, block=256
// ---------------------------------------------------------------------------
__global__ __launch_bounds__(256) void vq_cb_prep(
    const float* __restrict__ cb, _Float16* __restrict__ cbh,
    float* __restrict__ enorm)
{
    __shared__ float red[256];
    const int k = blockIdx.x;
    const int d = threadIdx.x;
    float v = cb[(size_t)k * DIMS + d];
    cbh[(size_t)k * DIMS + d] = (_Float16)v;
    red[d] = v * v;
    __syncthreads();
    #pragma unroll
    for (int s = 128; s > 0; s >>= 1) {
        if (d < s) red[d] += red[d + s];
        __syncthreads();
    }
    if (d == 0) enorm[k] = red[0];
}

// ---------------------------------------------------------------------------
// Kernel 2: z f32 -> f16 (vectorized x4). grid = NUMEL/4/256
// ---------------------------------------------------------------------------
__global__ __launch_bounds__(256) void vq_z_convert(
    const float* __restrict__ z, _Float16* __restrict__ zh)
{
    size_t i = (size_t)blockIdx.x * 256 + threadIdx.x;
    float4 v = ((const float4*)z)[i];
    v4h h = { (_Float16)v.x, (_Float16)v.y, (_Float16)v.z, (_Float16)v.w };
    ((v4h*)zh)[i] = h;
}

// ---------------------------------------------------------------------------
// Kernel 3: WMMA argmin.  block = 256 thr (8 waves), each wave owns 16 z-rows.
// A = codebook tile (16 entries x 32 K), B = z tile (32 K x 16 rows).
// Codebook double-buffered through LDS with async-to-LDS loads.
// grid = NROWS / 128 = 256
// ---------------------------------------------------------------------------
__global__ __launch_bounds__(256) void vq_argmin(
    const _Float16* __restrict__ zh,     // [NROWS][DIMS]
    const _Float16* __restrict__ cbh,    // [KENT][DIMS]
    const float*    __restrict__ enorm,  // [KENT]
    int*            __restrict__ outIdx) // [NROWS]
{
    __shared__ _Float16 cbLDS[2][CB_CHUNK * DIMS];   // 2 x 32 KB

    const int tid  = threadIdx.x;
    const int wave = tid >> 5;
    const int lane = tid & 31;
    const int l16  = lane & 15;
    const int hi   = lane >> 4;          // 0: K-low half, 1: K-high half

    const int rowBase = (blockIdx.x * 8 + wave) * 16;

    // B fragments: 16 z-rows x 256 K resident in registers for the whole sweep.
    // Lane l holds column n = l%16; 16 contiguous K starting at hi*16 + 32j.
    v16h bfrag[8];
    {
        const _Float16* zrow = zh + (size_t)(rowBase + l16) * DIMS;
        #pragma unroll
        for (int j = 0; j < 8; ++j)
            bfrag[j] = *(const v16h*)(zrow + j * 32 + hi * 16);
    }

    // Prime the double buffer with chunks 0 and 1.
    cb_issue_chunk(cbh,                             cbLDS[0], tid);
    cb_issue_chunk(cbh + (size_t)CB_CHUNK * DIMS,   cbLDS[1], tid);

    float minVal = 3.0e38f;
    int   minIdx = 0;

    for (int c = 0; c < NCHUNK; ++c) {
        // Ensure chunk c has landed (chunk c+1's 8 ops may stay in flight).
        if (c + 1 < NCHUNK) cb_wait_keep8(); else cb_wait_all();
        __syncthreads();
        const _Float16* buf = cbLDS[c & 1];

        #pragma unroll
        for (int t = 0; t < CB_CHUNK / 16; ++t) {
            // Preload all 8 A fragments so DS latency overlaps the WMMA chain.
            // A layout: lane l = row m = t*16 + l%16; halfs 0..15 = K j*32+hi*16,
            // contiguous.
            const _Float16* crow = buf + (t * 16 + l16) * DIMS;
            v16h afrag[8];
            #pragma unroll
            for (int j = 0; j < 8; ++j) {
                const _Float16* p0 = crow + j * 32 + hi * 16;
                v8h lo = *(const v8h*)p0;
                v8h hh = *(const v8h*)(p0 + 8);
                afrag[j] = __builtin_shufflevector(lo, hh,
                             0,1,2,3,4,5,6,7,8,9,10,11,12,13,14,15);
            }
            v8f acc = {};
            #pragma unroll
            for (int j = 0; j < 8; ++j)
                acc = __builtin_amdgcn_wmma_f32_16x16x32_f16(
                          false, afrag[j], false, bfrag[j], (short)0, acc,
                          false, false);

            // C layout: VGPR r holds codebook entry m = t*16 + hi*8 + r.
            const int gb = c * CB_CHUNK + t * 16 + hi * 8;
            float4 e0 = ((const float4*)(enorm + gb))[0];
            float4 e1 = ((const float4*)(enorm + gb))[1];
            float en[8] = { e0.x, e0.y, e0.z, e0.w, e1.x, e1.y, e1.z, e1.w };
            #pragma unroll
            for (int r = 0; r < 8; ++r) {
                float s = fmaf(-2.0f, acc[r], en[r]);   // ||e||^2 - 2 z.e
                if (s < minVal) { minVal = s; minIdx = gb + r; }
            }
        }

        __syncthreads();   // all waves done reading buf before overwriting it
        if (c + 2 < NCHUNK)
            cb_issue_chunk(cbh + (size_t)(c + 2) * CB_CHUNK * DIMS,
                           cbLDS[c & 1], tid);
    }

    // Combine the two half-wave candidate sets for each z-row.
    float oV = __shfl_xor(minVal, 16, 32);
    int   oI = __shfl_xor(minIdx, 16, 32);
    if (oV < minVal || (oV == minVal && oI < minIdx)) { minVal = oV; minIdx = oI; }
    if (hi == 0) outIdx[rowBase + l16] = minIdx;
}

// ---------------------------------------------------------------------------
// Kernel 4: gather quantized rows (exact f32 codebook) + per-block SSE partial
// block = 256 thr handles 4 rows (64 float4 per row). grid = NROWS/4 = 8192
// ---------------------------------------------------------------------------
__global__ __launch_bounds__(256) void vq_gather_loss(
    const float* __restrict__ z, const float* __restrict__ cb,
    const int* __restrict__ idxv, float* __restrict__ outq,
    float* __restrict__ partials)
{
    __shared__ float red[256];
    const int row = blockIdx.x * 4 + (threadIdx.x >> 6);
    const int dq  = threadIdx.x & 63;
    const int k   = idxv[row];
    float4 q  = ((const float4*)(cb + (size_t)k   * DIMS))[dq];
    float4 zv = ((const float4*)(z  + (size_t)row * DIMS))[dq];
    ((float4*)(outq + (size_t)row * DIMS))[dq] = q;
    float d0 = zv.x - q.x, d1 = zv.y - q.y, d2 = zv.z - q.z, d3 = zv.w - q.w;
    red[threadIdx.x] = d0*d0 + d1*d1 + d2*d2 + d3*d3;
    __syncthreads();
    #pragma unroll
    for (int s = 128; s > 0; s >>= 1) {
        if (threadIdx.x < s) red[threadIdx.x] += red[threadIdx.x + s];
        __syncthreads();
    }
    if (threadIdx.x == 0) partials[blockIdx.x] = red[0];
}

// ---------------------------------------------------------------------------
// Kernel 5: deterministic final reduction of 8192 partials -> vq_loss.
// ---------------------------------------------------------------------------
__global__ __launch_bounds__(256) void vq_finalize(
    const float* __restrict__ partials, float* __restrict__ outLoss)
{
    __shared__ float red[256];
    float s = 0.0f;
    for (int i = threadIdx.x; i < 8192; i += 256) s += partials[i];
    red[threadIdx.x] = s;
    __syncthreads();
    #pragma unroll
    for (int st = 128; st > 0; st >>= 1) {
        if (threadIdx.x < st) red[threadIdx.x] += red[threadIdx.x + st];
        __syncthreads();
    }
    // vq_loss = (1 + 0.25) * mean((z - quantized)^2)
    if (threadIdx.x == 0) outLoss[0] = 1.25f * red[0] / (float)NUMEL;
}

// ---------------------------------------------------------------------------
extern "C" void kernel_launch(void* const* d_in, const int* in_sizes, int n_in,
                              void* d_out, int out_size, void* d_ws, size_t ws_size,
                              hipStream_t stream)
{
    const float* z  = (const float*)d_in[0];   // 8,388,608 f32
    const float* cb = (const float*)d_in[1];   // 8192*256 f32
    float* outq = (float*)d_out;               // [NUMEL] quantized + [1] loss

    char* ws = (char*)d_ws;
    _Float16* zh       = (_Float16*)(ws);                           // 16 MB
    _Float16* cbh      = (_Float16*)(ws + 16777216);                // 4 MB
    float*    enorm    = (float*)   (ws + 16777216 + 4194304);      // 32 KB
    int*      idxv     = (int*)     (ws + 16777216 + 4194304 + 32768);
    float*    partials = (float*)   (ws + 16777216 + 4194304 + 32768 + 131072);

    vq_cb_prep    <<<KENT, 256, 0, stream>>>(cb, cbh, enorm);
    vq_z_convert  <<<NUMEL / 4 / 256, 256, 0, stream>>>(z, zh);
    vq_argmin     <<<NROWS / 128, 256, 0, stream>>>(zh, cbh, enorm, idxv);
    vq_gather_loss<<<NROWS / 4, 256, 0, stream>>>(z, cb, idxv, outq, partials);
    vq_finalize   <<<1, 256, 0, stream>>>(partials, outq + NUMEL);
}
